// MMDLoss_17282948399227
// MI455X (gfx1250) — matrix operand order repaired and checked by reading, not compiled, not run on hardware
//
#include <hip/hip_runtime.h>

typedef __attribute__((ext_vector_type(16))) _Float16 v16h;
typedef __attribute__((ext_vector_type(8)))  _Float16 v8h;
typedef __attribute__((ext_vector_type(8)))  float    v8f;

#define NROWS 8192
#define DDIM  256
#define TILE  128
#define LDS_STRIDE 264   // 256 halves + 8 pad -> 132 dwords/row, conflict-free frag reads
#define BETA  0.5f

#ifndef USE_ASYNC_LDS
#define USE_ASYNC_LDS 1   // gfx1250 GLOBAL_LOAD_ASYNC_TO_LDS_B128 staging path
#endif

union FragU { v16h v; v8h h[2]; };

// ---------------- kernel 1: f32 -> f16 convert + row norms (of converted values) ----
__global__ void mmd_convert_norms(const float* __restrict__ src,
                                  const float* __restrict__ tgt,
                                  _Float16* __restrict__ srcH,
                                  _Float16* __restrict__ tgtH,
                                  float* __restrict__ nsrc,
                                  float* __restrict__ ntgt) {
  __shared__ float red[256];
  int row = blockIdx.x;              // 0..16383 (source rows first, then target)
  int t = threadIdx.x;               // 0..255 == column
  const float* in; _Float16* outh; float* outn; int r;
  if (row < NROWS) { in = src; outh = srcH; outn = nsrc; r = row; }
  else             { in = tgt; outh = tgtH; outn = ntgt; r = row - NROWS; }
  float x = in[(size_t)r * DDIM + t];
  _Float16 h = (_Float16)x;
  outh[(size_t)r * DDIM + t] = h;
  float hf = (float)h;
  red[t] = hf * hf;
  __syncthreads();
  for (int s = 128; s > 0; s >>= 1) {
    if (t < s) red[t] += red[t + s];
    __syncthreads();
  }
  if (t == 0) outn[r] = red[0];
}

// ---------------- kernel 2: zero the 3 pair accumulators ---------------------------
__global__ void mmd_init_sums(float* sums) {
  if (threadIdx.x < 3) sums[threadIdx.x] = 0.0f;
}

// ---------------- kernel 3: fused WMMA Gram + exp + reduce -------------------------
// blockIdx.z: 0 = (src,src), 1 = (tgt,tgt), 2 = (src,tgt)
__global__ __launch_bounds__(256)
void mmd_gemm_exp(const _Float16* __restrict__ srcH,
                  const _Float16* __restrict__ tgtH,
                  const float* __restrict__ nsrc,
                  const float* __restrict__ ntgt,
                  float* __restrict__ sums) {
  extern __shared__ __align__(16) char smem[];
  _Float16* At = (_Float16*)smem;                 // 128 x 264 halves
  _Float16* Bt = At + TILE * LDS_STRIDE;          // 128 x 264 halves
  float* xn = (float*)(Bt + TILE * LDS_STRIDE);   // 128 row norms
  float* yn = xn + TILE;                          // 128 col norms
  __shared__ float wsum[8];

  const int z = blockIdx.z;
  const _Float16* X = (z == 1) ? tgtH : srcH;
  const _Float16* Y = (z == 0) ? srcH : tgtH;
  const float* nX = (z == 1) ? ntgt : nsrc;
  const float* nY = (z == 0) ? nsrc : ntgt;

  const int m0 = blockIdx.x * TILE;
  const int n0 = blockIdx.y * TILE;
  const int t  = threadIdx.x;

  // ---- stage full 128x256 f16 tiles of X and Y into LDS ----
  {
    const int r = t & 127;          // tile row
    const int hlf = t >> 7;         // which 128-half chunk of the row
    const _Float16* ga = X + (size_t)(m0 + r) * DDIM + hlf * 128;
    const _Float16* gb = Y + (size_t)(n0 + r) * DDIM + hlf * 128;
    _Float16* la = At + r * LDS_STRIDE + hlf * 128;
    _Float16* lb = Bt + r * LDS_STRIDE + hlf * 128;
#if USE_ASYNC_LDS
    // Direct global->LDS DMA (no VGPR staging), tracked by ASYNCcnt.
    unsigned ldsA = (unsigned)(size_t)la;   // generic LDS ptr low 32 bits = LDS byte addr
    unsigned ldsB = (unsigned)(size_t)lb;
#pragma unroll
    for (int c = 0; c < 16; ++c) {
      asm volatile("global_load_async_to_lds_b128 %0, %1, off"
                   :: "v"(ldsA + 16u * c), "v"((const void*)(ga + 8 * c))
                   : "memory");
      asm volatile("global_load_async_to_lds_b128 %0, %1, off"
                   :: "v"(ldsB + 16u * c), "v"((const void*)(gb + 8 * c))
                   : "memory");
    }
#else
    const uint4* gA4 = (const uint4*)ga;
    const uint4* gB4 = (const uint4*)gb;
    uint4* lA4 = (uint4*)la;
    uint4* lB4 = (uint4*)lb;
#pragma unroll
    for (int c = 0; c < 16; ++c) { lA4[c] = gA4[c]; lB4[c] = gB4[c]; }
#endif
  }
  if (t < 128) xn[t] = nX[m0 + t];
  else         yn[t - 128] = nY[n0 + (t - 128)];
#if USE_ASYNC_LDS
  asm volatile("s_wait_asynccnt 0x0" ::: "memory");
#endif
  __syncthreads();

  // ---- wave -> 2x4 grid of 16x16 subtiles ----
  const int w    = t >> 5;
  const int lane = t & 31;
  const int lm   = lane & 15;
  const int hi   = lane >> 4;
  const int wr   = (w >> 1) * 32;   // subtile-row base in tile (2 subtiles of 16)
  const int wc   = (w & 1) * 64;    // subtile-col base in tile (4 subtiles of 16)

  v8f acc[2][4];
#pragma unroll
  for (int i = 0; i < 2; ++i)
#pragma unroll
    for (int j = 0; j < 4; ++j) acc[i][j] = (v8f){};

#pragma unroll
  for (int k0 = 0; k0 < DDIM; k0 += 32) {
    FragU a[2];
#pragma unroll
    for (int i = 0; i < 2; ++i) {
      // A 16x32 f16 layout: lane lm+16*hi holds row lm, K = hi*8.. and 16+hi*8..
      const _Float16* pa = At + (wr + i * 16 + lm) * LDS_STRIDE + k0 + hi * 8;
      a[i].h[0] = *(const v8h*)pa;
      a[i].h[1] = *(const v8h*)(pa + 16);
    }
    FragU b[4];
#pragma unroll
    for (int j = 0; j < 4; ++j) {
      // B 32x16 f16 layout: lane holds col lm, K = hi*8.. and 16+hi*8..
      const _Float16* pb = Bt + (wc + j * 16 + lm) * LDS_STRIDE + k0 + hi * 8;
      b[j].h[0] = *(const v8h*)pb;
      b[j].h[1] = *(const v8h*)(pb + 16);
    }
#pragma unroll
    for (int i = 0; i < 2; ++i)
#pragma unroll
      for (int j = 0; j < 4; ++j)
        acc[i][j] = __builtin_amdgcn_wmma_f32_16x16x32_f16(
            false, a[i].v, false, b[j].v, (short)0, acc[i][j], false, false);
  }

  // ---- epilogue: d2 = x2 + y2 - 2c, clamp, exp, reduce ----
  float s = 0.0f;
#pragma unroll
  for (int i = 0; i < 2; ++i) {
#pragma unroll
    for (int j = 0; j < 4; ++j) {
      const float cy = yn[wc + j * 16 + lm];
#pragma unroll
      for (int p = 0; p < 8; ++p) {
        // C/D layout: VGPR p holds row m0+wr+i*16 + p + hi*8, col n0+wc+j*16+lm
        const float cx = xn[wr + i * 16 + p + hi * 8];
        float d2 = cx + cy - 2.0f * acc[i][j][p];
        d2 = fmaxf(d2, 0.0f);
        s += __expf(-BETA * d2);
      }
    }
  }
#pragma unroll
  for (int off = 16; off > 0; off >>= 1) s += __shfl_down(s, off, 32);
  if (lane == 0) wsum[w] = s;
  __syncthreads();
  if (t == 0) {
    float tot = 0.0f;
#pragma unroll
    for (int i = 0; i < 8; ++i) tot += wsum[i];
    atomicAdd(&sums[z], tot);
  }
}

// ---------------- kernel 4: combine ------------------------------------------------
__global__ void mmd_finalize(const float* __restrict__ sums, float* __restrict__ out) {
  const float inv = 1.0f / ((float)NROWS * (float)NROWS);
  out[0] = (sums[0] + sums[1] - 2.0f * sums[2]) * inv;
}

extern "C" void kernel_launch(void* const* d_in, const int* in_sizes, int n_in,
                              void* d_out, int out_size, void* d_ws, size_t ws_size,
                              hipStream_t stream) {
  (void)in_sizes; (void)n_in; (void)out_size; (void)ws_size;
  const float* src = (const float*)d_in[0];
  const float* tgt = (const float*)d_in[1];
  float* out = (float*)d_out;

  char* ws = (char*)d_ws;
  const size_t HBYTES = (size_t)NROWS * DDIM * sizeof(_Float16);  // 4 MiB
  _Float16* srcH = (_Float16*)(ws);
  _Float16* tgtH = (_Float16*)(ws + HBYTES);
  float* nsrc = (float*)(ws + 2 * HBYTES);
  float* ntgt = (float*)(ws + 2 * HBYTES + NROWS * sizeof(float));
  float* sums = (float*)(ws + 2 * HBYTES + 2 * NROWS * sizeof(float));

  mmd_convert_norms<<<2 * NROWS, 256, 0, stream>>>(src, tgt, srcH, tgtH, nsrc, ntgt);
  mmd_init_sums<<<1, 32, 0, stream>>>(sums);

  const size_t shmem = (size_t)2 * TILE * LDS_STRIDE * sizeof(_Float16)
                     + (size_t)2 * TILE * sizeof(float);   // ~133 KB: 2 blocks/WGP
  dim3 grid(NROWS / TILE, NROWS / TILE, 3);
  mmd_gemm_exp<<<grid, 256, shmem, stream>>>(srcH, tgtH, nsrc, ntgt, sums);

  mmd_finalize<<<1, 1, 0, stream>>>(sums, out);
}
